// EquivariantNet_59820304498797
// MI455X (gfx1250) — matrix operand order repaired and checked by reading, not compiled, not run on hardware
//
#include <hip/hip_runtime.h>
#include <hip/hip_bf16.h>
#include <math.h>

// ---------------------------------------------------------------------------
// CDNA5 / gfx1250 implementation of the SE(3)-equivariant GNN forward pass.
// Heavy per-edge contractions (h=16, i=16) run on v_wmma_f32_16x16x32_bf16
// with 16-edge M-tiles; K=16 is zero-padded to 32. Columns of each WMMA are
// the 'i' channel so the per-edge epilogue is a 16-lane butterfly reduction.
// Weights are pre-packed once per launch into WMMA-ready bf16 B-fragments so
// the steady-state inner loop is: 32B vector load + v_wmma + reduction.
// bf16 conversions use native casts (clang legalizes to hardware cvt, RNE).
// ---------------------------------------------------------------------------

typedef __attribute__((ext_vector_type(16))) __bf16 vbf16;
typedef __attribute__((ext_vector_type(8)))  __bf16 v8bf;
typedef __attribute__((ext_vector_type(8)))  float  v8f;

__device__ __forceinline__ __bf16 f2bf(float f) { return (__bf16)f; }
__device__ __forceinline__ float  bf2f(__bf16 b) { return (float)b; }
__device__ __forceinline__ vbf16 zero_vbf16() {
  vbf16 z;
#pragma unroll
  for (int j = 0; j < 16; ++j) z[j] = __builtin_bit_cast(__bf16, (unsigned short)0);
  return z;
}
// sum across the 16 lanes of each half-wave (xor masks < 16 never cross halves)
__device__ __forceinline__ float red16(float v) {
  v += __shfl_xor(v, 1, 32);
  v += __shfl_xor(v, 2, 32);
  v += __shfl_xor(v, 4, 32);
  v += __shfl_xor(v, 8, 32);
  return v;
}

// ---------------------------------------------------------------------------
// Weight pre-pack: B fragments stored as bf16[group][col n(16)][k element j(16)]
// group map: [0,4)            : H weights, p = g          B[k=f][n=h]
//            [4, 4+3*KV)      : branch br=(g-4)/KV, o=%KV B[k=h][n=i]
//            [4+3*KV, 4+6*KV) : branch3 (o,f)=((g')/3,%3) B[k=h][n=i]
// ---------------------------------------------------------------------------
template <int KV>
__global__ void pack_weights_kernel(const float* __restrict__ w1,   // (4,16,16)
                                    const float* __restrict__ w2n,  // (3,16,KV,16)
                                    const float* __restrict__ w2v,  // (16,KV,16,3)
                                    __bf16* __restrict__ outp) {
  constexpr int NG = 4 + 6 * KV;
  const int idx = blockIdx.x * blockDim.x + threadIdx.x;
  if (idx >= NG * 256) return;
  const int g = idx >> 8;
  const int n = (idx >> 4) & 15;   // column
  const int j = idx & 15;          // k element
  float v;
  if (g < 4) {
    v = w1[(g * 16 + j) * 16 + n];
  } else if (g < 4 + 3 * KV) {
    const int br = (g - 4) / KV, o = (g - 4) % KV;
    v = w2n[((br * 16 + j) * KV + o) * 16 + n];
  } else {
    const int gg = g - 4 - 3 * KV;
    const int o = gg / 3, f = gg % 3;
    v = w2v[((j * KV + o) * 16 + n) * 3 + f];
  }
  outp[idx] = f2bf(v);
}

// ---------------------------------------------------------------------------
// conv_edges: per 16-edge tile (one wave):
//   H[p] = relu(ef @ w1[p])                          (WMMA, A=ef tile)
//   R    = H[b] @ w2[b]                              (WMMA, B=packed weights)
//   out  = sum_i R * tmp_coef                        (butterfly reduction)
// branches: 0:(H0,b00,x0)->m0  2:(H2,b10,x1)->m0  1:(H1,b01,x0)->m1
//           3:(H3,b11,x1)->m1
// KV = 16 for the attention kv convs, 8 for the final conv.
// ---------------------------------------------------------------------------
template <int KV>
__global__ __launch_bounds__(128) void conv_edges_kernel(
    const float* __restrict__ ef,                       // (E,16)
    const float* __restrict__ b00, const float* __restrict__ b01,
    const float* __restrict__ b10, const float* __restrict__ b11,
    const __bf16* __restrict__ wfrag,                   // packed B fragments
    const float* __restrict__ x0,                       // (N,16)
    const float* __restrict__ x1,                       // (N,16,3)
    const int*   __restrict__ esrc,
    float* __restrict__ m0,                             // (E,KV)
    float* __restrict__ m1,                             // (E,KV,3)
    int E)
{
  constexpr int WAVES = 4;
  __shared__ float  s_x0 [WAVES][16][16];       // x0[src] per edge
  __shared__ float  s_t10[WAVES][16][16];       // sum_q b10[q]*x1[i,q]
  __shared__ __bf16 s_t11[WAVES][16][16][3][3]; // [e][i][p][f]
  __shared__ float  s_b00[WAVES][16];
  __shared__ float  s_b01[WAVES][16][3];
  __shared__ __attribute__((aligned(32))) __bf16 s_H[WAVES][4][16][16]; // relu'd H
  __shared__ float  s_ac0[WAVES][16][16];       // m0 accumulator [e][o]
  __shared__ float  s_ac1[WAVES][16][16][3];    // m1 accumulator [e][o][p]

  const int lane = threadIdx.x & 31;
  const int wv   = threadIdx.x >> 5;
  const int tile = blockIdx.x * WAVES + wv;     // grid sized exactly: E % 64 == 0
  const int e0   = tile * 16;
  const int row  = lane & 15;                   // M-row / N-col / i index
  const int half = lane >> 4;                   // k-half selector
  const vbf16 zb = zero_vbf16();
  const vbf16* __restrict__ wf = (const vbf16*)wfrag;   // wf[g*16 + n]

  // ---- phase 1: gather per-edge operands into LDS ----
  {
    const int e   = e0 + row;
    const int src = esrc[e];
    if (half == 0) {
      s_b00[wv][row]    = b00[e];
      s_b01[wv][row][0] = b01[e * 3 + 0];
      s_b01[wv][row][1] = b01[e * 3 + 1];
      s_b01[wv][row][2] = b01[e * 3 + 2];
    }
    const float g0 = b10[e * 3 + 0], g1 = b10[e * 3 + 1], g2 = b10[e * 3 + 2];
#pragma unroll
    for (int ii = 0; ii < 8; ++ii) {
      const int i = half * 8 + ii;
      s_x0[wv][row][i] = x0[src * 16 + i];
      const float xq0 = x1[(src * 16 + i) * 3 + 0];
      const float xq1 = x1[(src * 16 + i) * 3 + 1];
      const float xq2 = x1[(src * 16 + i) * 3 + 2];
      s_t10[wv][row][i] = g0 * xq0 + g1 * xq1 + g2 * xq2;
#pragma unroll
      for (int p = 0; p < 3; ++p)
#pragma unroll
        for (int f = 0; f < 3; ++f) {
          const float c0 = b11[e * 27 + p * 9 + 0 * 3 + f];
          const float c1 = b11[e * 27 + p * 9 + 1 * 3 + f];
          const float c2 = b11[e * 27 + p * 9 + 2 * 3 + f];
          s_t11[wv][row][i][p][f] = f2bf(c0 * xq0 + c1 * xq1 + c2 * xq2);
        }
    }
  }

  // A fragment of edge_feat tile: lane holds row (edge); half 0 holds k 0..15, half 1 zero (pad K->32)
  vbf16 a_ef = zb;
#pragma unroll
  for (int j = 0; j < 16; ++j) if (half == 0) a_ef[j] = f2bf(ef[(e0 + row) * 16 + j]);

  // zero accumulators
  for (int idx = lane; idx < 16 * 16; idx += 32) s_ac0[wv][idx >> 4][idx & 15] = 0.f;
  for (int idx = lane; idx < 16 * 16 * 3; idx += 32) (&s_ac1[wv][0][0][0])[idx] = 0.f;
  __syncthreads();

  // ---- phase 2: H[p] = relu(ef @ w1[p]) via WMMA ----
#pragma unroll
  for (int p = 0; p < 4; ++p) {
    const vbf16 b = (half == 0) ? wf[p * 16 + row] : zb;
    v8f c = {};
    c = __builtin_amdgcn_wmma_f32_16x16x32_bf16(false, a_ef, false, b, (short)0,
                                                c, false, false);
#pragma unroll
    for (int r = 0; r < 8; ++r)  // D: lane = col h, rows r (+8 for hi half)
      s_H[wv][p][half * 8 + r][row] = f2bf(fmaxf(c[r], 0.f));
  }
  __syncthreads();

  // A fragment from relu'd H (K=16 padded to 32); one 32B LDS vector read on half 0
  auto loadA = [&](int br) {
    vbf16 a = zb;
    if (half == 0) a = *(const vbf16*)&s_H[wv][br][row][0];
    return a;
  };

  // ---- phase 3: R GEMMs + reduction epilogues ----
  // f=1 branches (0 -> m0 w/ b00*x0 ; 2 -> m0 w/ t10 ; 1 -> m1 w/ x0 then *b01)
  {
    const vbf16 a0 = loadA(0);
    for (int g = 0; g < KV; ++g) {
      const vbf16 b = (half == 0) ? wf[(4 + 0 * KV + g) * 16 + row] : zb;
      v8f d = {};
      d = __builtin_amdgcn_wmma_f32_16x16x32_bf16(false, a0, false, b, (short)0,
                                                  d, false, false);
#pragma unroll
      for (int r = 0; r < 8; ++r) {
        const int e = half * 8 + r;
        const float v = red16(d[r] * (s_b00[wv][e] * s_x0[wv][e][row]));
        if (row == 0) s_ac0[wv][e][g] += v;
      }
    }
  }
  {
    const vbf16 a2 = loadA(2);
    for (int g = 0; g < KV; ++g) {
      const vbf16 b = (half == 0) ? wf[(4 + 2 * KV + g) * 16 + row] : zb;
      v8f d = {};
      d = __builtin_amdgcn_wmma_f32_16x16x32_bf16(false, a2, false, b, (short)0,
                                                  d, false, false);
#pragma unroll
      for (int r = 0; r < 8; ++r) {
        const int e = half * 8 + r;
        const float v = red16(d[r] * s_t10[wv][e][row]);
        if (row == 0) s_ac0[wv][e][g] += v;
      }
    }
  }
  {
    const vbf16 a1 = loadA(1);
    for (int g = 0; g < KV; ++g) {
      const vbf16 b = (half == 0) ? wf[(4 + 1 * KV + g) * 16 + row] : zb;
      v8f d = {};
      d = __builtin_amdgcn_wmma_f32_16x16x32_bf16(false, a1, false, b, (short)0,
                                                  d, false, false);
#pragma unroll
      for (int r = 0; r < 8; ++r) {
        const int e = half * 8 + r;
        const float v = red16(d[r] * s_x0[wv][e][row]);
        if (row == 0) {
          s_ac1[wv][e][g][0] += v * s_b01[wv][e][0];
          s_ac1[wv][e][g][1] += v * s_b01[wv][e][1];
          s_ac1[wv][e][g][2] += v * s_b01[wv][e][2];
        }
      }
    }
  }
  // branch 3 (f=3): groups are (o,f), lanes = i; epilogue contracts (i,f)
  {
    const vbf16 a3 = loadA(3);
    for (int o = 0; o < KV; ++o) {
#pragma unroll
      for (int f = 0; f < 3; ++f) {
        const vbf16 b = (half == 0) ? wf[(4 + 3 * KV + o * 3 + f) * 16 + row] : zb;
        v8f d = {};
        d = __builtin_amdgcn_wmma_f32_16x16x32_bf16(false, a3, false, b,
                                                    (short)0, d, false, false);
#pragma unroll
        for (int r = 0; r < 8; ++r) {
          const int e = half * 8 + r;
          const float rv = d[r];
          const float v0 = red16(rv * bf2f(s_t11[wv][e][row][0][f]));
          const float v1 = red16(rv * bf2f(s_t11[wv][e][row][1][f]));
          const float v2 = red16(rv * bf2f(s_t11[wv][e][row][2][f]));
          if (row == 0) {
            s_ac1[wv][e][o][0] += v0;
            s_ac1[wv][e][o][1] += v1;
            s_ac1[wv][e][o][2] += v2;
          }
        }
      }
    }
  }
  __syncthreads();

  // ---- phase 4: write tile outputs ----
  for (int idx = lane; idx < 16 * KV; idx += 32) {
    const int e = idx / KV, o = idx % KV;
    m0[(size_t)(e0 + e) * KV + o] = s_ac0[wv][e][o];
  }
  for (int idx = lane; idx < 16 * KV * 3; idx += 32) {
    const int e = idx / (KV * 3), rem = idx % (KV * 3);
    m1[(size_t)(e0 + e) * KV * 3 + rem] = s_ac1[wv][e][rem / 3][rem % 3];
  }
}

// ---------------------------------------------------------------------------
// attention pipeline
// ---------------------------------------------------------------------------
__global__ void q_kernel(const float* __restrict__ x0, const float* __restrict__ x1,
                         const float* __restrict__ wq, float* __restrict__ q0,
                         float* __restrict__ q1, int N) {
  const int idx = blockIdx.x * blockDim.x + threadIdx.x;
  if (idx >= N * 8) return;
  const int n = idx >> 3, d = idx & 7;
  const float* wq0 = wq;         // (16,8)
  const float* wq1 = wq + 128;   // (16,8)
  float a0 = 0.f, ax = 0.f, ay = 0.f, az = 0.f;
  for (int c = 0; c < 16; ++c) {
    const float w0 = wq0[c * 8 + d], w1v = wq1[c * 8 + d];
    a0 += x0[n * 16 + c] * w0;
    ax += x1[(n * 16 + c) * 3 + 0] * w1v;
    ay += x1[(n * 16 + c) * 3 + 1] * w1v;
    az += x1[(n * 16 + c) * 3 + 2] * w1v;
  }
  q0[idx] = a0;
  q1[idx * 3 + 0] = ax; q1[idx * 3 + 1] = ay; q1[idx * 3 + 2] = az;
}

__global__ void zmax_kernel(const float* __restrict__ kv0, const float* __restrict__ kv1,
                            const float* __restrict__ q0, const float* __restrict__ q1,
                            const int* __restrict__ edst, float* __restrict__ z,
                            unsigned* __restrict__ zmax, int E, float scale) {
  const int e = blockIdx.x * blockDim.x + threadIdx.x;
  if (e >= E) return;
  const int n = edst[e];
  float zh[4] = {0.f, 0.f, 0.f, 0.f};
  for (int d = 0; d < 8; ++d) {
    float t = q0[n * 8 + d] * kv0[e * 16 + d];
    for (int m = 0; m < 3; ++m)
      t += q1[(n * 8 + d) * 3 + m] * kv1[(e * 16 + d) * 3 + m];
    zh[d >> 1] += t;
  }
  for (int h = 0; h < 4; ++h) {
    const float zv = zh[h] * scale;
    z[e * 4 + h] = zv;
    const unsigned bits = __builtin_bit_cast(unsigned, zv);
    const unsigned enc = (bits & 0x80000000u) ? ~bits : (bits | 0x80000000u);
    atomicMax(&zmax[n * 4 + h], enc);
  }
}

__global__ void expsum_kernel(const float* __restrict__ z, const unsigned* __restrict__ zmax,
                              const int* __restrict__ edst, float* __restrict__ ez,
                              float* __restrict__ zsum, int E) {
  const int e = blockIdx.x * blockDim.x + threadIdx.x;
  if (e >= E) return;
  const int n = edst[e];
  for (int h = 0; h < 4; ++h) {
    const unsigned enc = zmax[n * 4 + h];
    const unsigned bits = (enc & 0x80000000u) ? (enc & 0x7FFFFFFFu) : ~enc;
    float m = __builtin_bit_cast(float, bits);
    if (!__builtin_isfinite(m)) m = 0.f;          // matches reference where()
    const float ex = __expf(z[e * 4 + h] - m);
    ez[e * 4 + h] = ex;
    atomicAdd(&zsum[n * 4 + h], ex);
  }
}

__global__ void agg_kernel(const float* __restrict__ kv0, const float* __restrict__ kv1,
                           const float* __restrict__ ez, const float* __restrict__ zsum,
                           const int* __restrict__ edst, float* __restrict__ o0,
                           float* __restrict__ o1, int E) {
  const int e = blockIdx.x * blockDim.x + threadIdx.x;
  if (e >= E) return;
  const int n = edst[e];
  float alpha[4];
  for (int h = 0; h < 4; ++h) alpha[h] = ez[e * 4 + h] / (zsum[n * 4 + h] + 1e-9f);
  for (int d = 0; d < 8; ++d) {
    const float al = alpha[d >> 1];
    atomicAdd(&o0[n * 8 + d], al * kv0[e * 16 + 8 + d]);
    for (int m = 0; m < 3; ++m)
      atomicAdd(&o1[(n * 8 + d) * 3 + m], al * kv1[(e * 16 + 8 + d) * 3 + m]);
  }
}

__global__ void update_kernel(const float* __restrict__ o0, const float* __restrict__ o1,
                              const float* __restrict__ x0, const float* __restrict__ x1,
                              const float* __restrict__ wproj, const float* __restrict__ gamma,
                              const float* __restrict__ beta, float* __restrict__ nx0,
                              float* __restrict__ nx1, int N) {
  const int n = blockIdx.x * blockDim.x + threadIdx.x;
  if (n >= N) return;
  const float* wp0 = wproj;            // (24,16)
  const float* wp1 = wproj + 24 * 16;  // (24,16)
  float y0[16], y1[16][3];
  for (int c = 0; c < 16; ++c) {
    float a = 0.f;
    for (int k = 0; k < 8; ++k)  a += o0[n * 8 + k]  * wp0[k * 16 + c];
    for (int k = 0; k < 16; ++k) a += x0[n * 16 + k] * wp0[(8 + k) * 16 + c];
    y0[c] = a;
    for (int m = 0; m < 3; ++m) {
      float b = 0.f;
      for (int k = 0; k < 8; ++k)  b += o1[(n * 8 + k) * 3 + m]  * wp1[k * 16 + c];
      for (int k = 0; k < 16; ++k) b += x1[(n * 16 + k) * 3 + m] * wp1[(8 + k) * 16 + c];
      y1[c][m] = b;
    }
  }
  { // SE(3) norm, degree 0
    float nv[16], mu = 0.f;
    for (int c = 0; c < 16; ++c) { nv[c] = sqrtf(y0[c] * y0[c] + 1e-12f); mu += nv[c]; }
    mu *= (1.f / 16.f);
    float var = 0.f;
    for (int c = 0; c < 16; ++c) { const float d = nv[c] - mu; var += d * d; }
    var *= (1.f / 16.f);
    const float inv = rsqrtf(var + 1e-5f);
    for (int c = 0; c < 16; ++c) {
      const float ln = (nv[c] - mu) * inv * gamma[c] + beta[c];
      nx0[n * 16 + c] = y0[c] * (fmaxf(ln, 0.f) / (nv[c] + 1e-3f));
    }
  }
  { // SE(3) norm, degree 1
    float nv[16], mu = 0.f;
    for (int c = 0; c < 16; ++c) {
      const float s = y1[c][0] * y1[c][0] + y1[c][1] * y1[c][1] + y1[c][2] * y1[c][2];
      nv[c] = sqrtf(s + 1e-12f); mu += nv[c];
    }
    mu *= (1.f / 16.f);
    float var = 0.f;
    for (int c = 0; c < 16; ++c) { const float d = nv[c] - mu; var += d * d; }
    var *= (1.f / 16.f);
    const float inv = rsqrtf(var + 1e-5f);
    for (int c = 0; c < 16; ++c) {
      const float ln = (nv[c] - mu) * inv * gamma[16 + c] + beta[16 + c];
      const float r = fmaxf(ln, 0.f) / (nv[c] + 1e-3f);
      for (int m = 0; m < 3; ++m)
        nx1[(n * 16 + c) * 3 + m] = y1[c][m] * r;
    }
  }
}

// ---------------------------------------------------------------------------
// final conv aggregation: out[n] = segment_sum(m) + x @ wself
// out layout: [n][0..7] = y0, [n][8 + o*3 + m] = y1
// ---------------------------------------------------------------------------
__global__ void fc_scatter_kernel(const float* __restrict__ m0, const float* __restrict__ m1,
                                  const int* __restrict__ edst, float* __restrict__ out, int E) {
  const int e = blockIdx.x * blockDim.x + threadIdx.x;
  if (e >= E) return;
  const int n = edst[e];
  for (int o = 0; o < 8; ++o) atomicAdd(&out[n * 32 + o], m0[(size_t)e * 8 + o]);
  for (int o = 0; o < 8; ++o)
    for (int m = 0; m < 3; ++m)
      atomicAdd(&out[n * 32 + 8 + o * 3 + m], m1[((size_t)e * 8 + o) * 3 + m]);
}

__global__ void fc_self_kernel(const float* __restrict__ x0, const float* __restrict__ x1,
                               const float* __restrict__ wself, float* __restrict__ out, int N) {
  const int n = blockIdx.x * blockDim.x + threadIdx.x;
  if (n >= N) return;
  for (int o = 0; o < 8; ++o) {
    float a = 0.f;
    for (int c = 0; c < 16; ++c) a += x0[n * 16 + c] * wself[c * 8 + o];
    out[n * 32 + o] += a;
  }
  for (int o = 0; o < 8; ++o)
    for (int m = 0; m < 3; ++m) {
      float b = 0.f;
      for (int c = 0; c < 16; ++c) b += x1[(n * 16 + c) * 3 + m] * wself[128 + c * 8 + o];
      out[n * 32 + 8 + o * 3 + m] += b;
    }
}

// ---------------------------------------------------------------------------
extern "C" void kernel_launch(void* const* d_in, const int* in_sizes, int n_in,
                              void* d_out, int out_size, void* d_ws, size_t ws_size,
                              hipStream_t stream) {
  const float* x0_in  = (const float*)d_in[0];
  const float* x1_in  = (const float*)d_in[1];
  const float* ef     = (const float*)d_in[2];
  const float* b00    = (const float*)d_in[3];
  const float* b01    = (const float*)d_in[4];
  const float* b10    = (const float*)d_in[5];
  const float* b11    = (const float*)d_in[6];
  const float* kv_w1  = (const float*)d_in[7];
  const float* kv_w2n = (const float*)d_in[8];
  const float* kv_w2v = (const float*)d_in[9];
  const float* wq     = (const float*)d_in[10];
  const float* wproj  = (const float*)d_in[11];
  const float* gamma  = (const float*)d_in[12];
  const float* beta   = (const float*)d_in[13];
  const float* fc_w1  = (const float*)d_in[14];
  const float* fc_w2n = (const float*)d_in[15];
  const float* fc_w2v = (const float*)d_in[16];
  const float* wself  = (const float*)d_in[17];
  const int*   esrc   = (const int*)d_in[18];
  const int*   edst   = (const int*)d_in[19];
  float* out = (float*)d_out;

  const int N = in_sizes[0] / 16;   // 8192
  const int E = in_sizes[18];       // 131072 (E % 64 == 0)

  // workspace carve-up (floats); keep 32B alignment for bf16 fragment caches
  float* ws = (float*)d_ws;
  size_t off = 0;
  auto alloc = [&](size_t nf) {
    off = (off + 7) & ~(size_t)7;   // 32-byte align
    float* p = ws + off; off += nf; return p;
  };
  float* kvm0 = alloc((size_t)E * 16);
  float* kvm1 = alloc((size_t)E * 48);
  float* q0   = alloc((size_t)N * 8);
  float* q1   = alloc((size_t)N * 24);
  float* z    = alloc((size_t)E * 4);
  float* ez   = alloc((size_t)E * 4);
  float* zmxf = alloc((size_t)N * 4);
  float* zsum = alloc((size_t)N * 4);
  float* o0   = alloc((size_t)N * 8);
  float* o1   = alloc((size_t)N * 24);
  float* xa0  = alloc((size_t)N * 16);
  float* xa1  = alloc((size_t)N * 48);
  float* xb0  = alloc((size_t)N * 16);
  float* xb1  = alloc((size_t)N * 48);
  // packed bf16 weight fragments: (4 + 6*KV) groups * 256 bf16 each
  constexpr int NG16 = 4 + 6 * 16, NG8 = 4 + 6 * 8;
  __bf16* wpk0 = (__bf16*)alloc((size_t)(NG16 * 256 + 1) / 2);
  __bf16* wpk1 = (__bf16*)alloc((size_t)(NG16 * 256 + 1) / 2);
  __bf16* wpkf = (__bf16*)alloc((size_t)(NG8 * 256 + 1) / 2);
  unsigned* zmax = (unsigned*)zmxf;

  const float scale = 0.35355339059327373f;  // 1/sqrt((ckv/HEADS)*4) = 1/sqrt(8)

  const int convBlocks = E / 64;             // 4 waves/block * 16 edges/wave
  const int eBlocks = (E + 255) / 256;
  const int nBlocks = (N + 127) / 128;
  const int qBlocks = (N * 8 + 255) / 256;

  // ---- pre-pack all conv weights into WMMA B-fragment layout ----
  pack_weights_kernel<16><<<(NG16 * 256 + 255) / 256, 256, 0, stream>>>(
      kv_w1, kv_w2n, kv_w2v, wpk0);
  pack_weights_kernel<16><<<(NG16 * 256 + 255) / 256, 256, 0, stream>>>(
      kv_w1 + 1024, kv_w2n + 12288, kv_w2v + 12288, wpk1);
  pack_weights_kernel<8><<<(NG8 * 256 + 255) / 256, 256, 0, stream>>>(
      fc_w1, fc_w2n, fc_w2v, wpkf);

  const float* cx0 = x0_in;
  const float* cx1 = x1_in;
  float* outs0[2] = {xa0, xb0};
  float* outs1[2] = {xa1, xb1};
  __bf16* wpks[2] = {wpk0, wpk1};

  for (int l = 0; l < 2; ++l) {
    conv_edges_kernel<16><<<convBlocks, 128, 0, stream>>>(
        ef, b00, b01, b10, b11, wpks[l], cx0, cx1, esrc, kvm0, kvm1, E);

    q_kernel<<<qBlocks, 256, 0, stream>>>(cx0, cx1, wq + (size_t)l * 256, q0, q1, N);

    hipMemsetAsync(zmax, 0, (size_t)N * 4 * sizeof(unsigned), stream);
    hipMemsetAsync(zsum, 0, (size_t)N * 4 * sizeof(float), stream);
    hipMemsetAsync(o0, 0, (size_t)N * 8 * sizeof(float), stream);
    hipMemsetAsync(o1, 0, (size_t)N * 24 * sizeof(float), stream);

    zmax_kernel<<<eBlocks, 256, 0, stream>>>(kvm0, kvm1, q0, q1, edst, z, zmax, E, scale);
    expsum_kernel<<<eBlocks, 256, 0, stream>>>(z, zmax, edst, ez, zsum, E);
    agg_kernel<<<eBlocks, 256, 0, stream>>>(kvm0, kvm1, ez, zsum, edst, o0, o1, E);

    update_kernel<<<nBlocks, 128, 0, stream>>>(
        o0, o1, cx0, cx1, wproj + (size_t)l * 768,
        gamma + (size_t)l * 32, beta + (size_t)l * 32, outs0[l], outs1[l], N);

    cx0 = outs0[l];
    cx1 = outs1[l];
  }

  // final conv (Co = 8), reuse kvm buffers
  conv_edges_kernel<8><<<convBlocks, 128, 0, stream>>>(
      ef, b00, b01, b10, b11, wpkf, cx0, cx1, esrc, kvm0, kvm1, E);

  hipMemsetAsync(out, 0, (size_t)out_size * sizeof(float), stream);
  fc_scatter_kernel<<<eBlocks, 256, 0, stream>>>(kvm0, kvm1, edst, out, E);
  fc_self_kernel<<<nBlocks, 128, 0, stream>>>(cx0, cx1, wself, out, N);
}
